// TinyGPT_66752381714782
// MI455X (gfx1250) — compile-verified, matrix operand order checked by reading, not run on hardware
//
#include <hip/hip_runtime.h>
#include <hip/hip_bf16.h>
#include <math.h>

// ---------------------------------------------------------------------------
// TinyGPT forward for MI455X (gfx1250, wave32, WMMA).
// Heavy math: V_WMMA_F32_16X16X32_BF16 (bf16 in, f32 accum).
// Global->LDS staging: GLOBAL_LOAD_ASYNC_TO_LDS_B128 (ASYNCcnt), double-buffered.
// NOTE: all callers guarantee M % BM == 0 and K % BK == 0; only N may be ragged
// (logits GEMM, N=50257), handled by a block-uniform full/partial path select.
// ---------------------------------------------------------------------------

typedef __bf16 bf16;
typedef __attribute__((ext_vector_type(16))) __bf16 v16bf;
typedef __attribute__((ext_vector_type(8)))  float  v8f;

#define BM 128
#define BN 128
#define BK 32
#define LDSROW 40   // bf16 elems per LDS tile row: 80 bytes -> 16B aligned, bank-spread

union FragU { uint4 q[2]; v16bf v; };
union Pack8 { uint4 q; bf16 e[8]; };

// wave-relative LDS byte offset of a generic pointer into __shared__ memory
static __device__ __forceinline__ unsigned lds_off(const void* p) {
  return (unsigned)(size_t)p;   // low 32 bits of flat LDS address == LDS offset
}

static __device__ __forceinline__ void async_copy16(const bf16* gsrc, bf16* ldst) {
  asm volatile("global_load_async_to_lds_b128 %0, %1, off"
               :: "v"(lds_off(ldst)), "v"((unsigned long long)gsrc)
               : "memory");
}
static __device__ __forceinline__ void wait_async0() {
  asm volatile("s_wait_asynccnt 0x0" ::: "memory");
}

// ---------------------------------------------------------------------------
// Generic bf16 GEMM:  out[M,N] = (A[M,K] * B + bias) * scale
//   B_IS_KN=false : B is [N,K] row-major (pre-transposed weights / k-matrix)
//   B_IS_KN=true  : B is [K,N] row-major (v-matrix), transposed via LDS
// causal=1: skip blocks entirely above the diagonal (for S = q*k^T).
// Writes f32 (outF) or bf16 (outB).
// ---------------------------------------------------------------------------
template<bool B_IS_KN>
__global__ __launch_bounds__(256) void gemm_bf16_wmma(
    const bf16* __restrict__ A, const bf16* __restrict__ B,
    const float* __restrict__ bias,
    float* __restrict__ outF, bf16* __restrict__ outB,
    int M, int N, int K, int lda, int ldb, int ldc,
    float scale, int causal)
{
  const int m0 = blockIdx.y * BM;
  const int n0 = blockIdx.x * BN;
  if (causal && (m0 + BM - 1) < n0) return;   // fully masked block

  __shared__ __align__(16) bf16 As[2][BM * LDSROW];
  __shared__ __align__(16) bf16 Bs[2][BN * LDSROW];

  const int tid  = threadIdx.x;
  const int lane = tid & 31;
  const int wave = tid >> 5;
  const int wm   = wave & 3;     // 4 waves along M
  const int wn   = wave >> 2;    // 2 waves along N
  const int lm   = lane & 15;
  const int half = lane >> 4;

  const bool nFull = (n0 + BN) <= N;   // block-uniform: B tile fully in range?

  // A tile [BM x BK] (M always a multiple of BM -> branch-free async DMA)
  auto stageA = [&](int k0, int buf) {
#pragma unroll
    for (int j = 0; j < 2; ++j) {
      int c   = tid * 2 + j;          // 0..511
      int row = c >> 2;               // 0..127
      int col = (c & 3) * 8;          // 0,8,16,24
      async_copy16(A + (size_t)(m0 + row) * lda + k0 + col,
                   &As[buf][row * LDSROW + col]);
    }
  };
  // B tile [BN x BK] in N-major form
  auto stageB = [&](int k0, int buf) {
    if (B_IS_KN) {
      // B is [K,N] (N, K exact multiples in all uses): transpose via registers
#pragma unroll
      for (int j = 0; j < 2; ++j) {
        int c  = tid * 2 + j;
        int kk = c >> 4;              // 0..31
        int nc = (c & 15) * 8;        // 0..120
        Pack8 u;
        u.q = *(const uint4*)(B + (size_t)(k0 + kk) * ldb + n0 + nc);
#pragma unroll
        for (int e = 0; e < 8; ++e) Bs[buf][(nc + e) * LDSROW + kk] = u.e[e];
      }
    } else if (nFull) {
      // fast path: branch-free async DMA
#pragma unroll
      for (int j = 0; j < 2; ++j) {
        int c   = tid * 2 + j;
        int row = c >> 2;
        int col = (c & 3) * 8;
        async_copy16(B + (size_t)(n0 + row) * ldb + k0 + col,
                     &Bs[buf][row * LDSROW + col]);
      }
    } else {
      // ragged-N path (last logits block column only)
#pragma unroll
      for (int j = 0; j < 2; ++j) {
        int c   = tid * 2 + j;
        int row = c >> 2;
        int col = (c & 3) * 8;
        bf16* dst = &Bs[buf][row * LDSROW + col];
        int gr = n0 + row;
        if (gr < N) {
          async_copy16(B + (size_t)gr * ldb + k0 + col, dst);
        } else {
          uint4 z; z.x = z.y = z.z = z.w = 0u;
          *(uint4*)dst = z;
        }
      }
    }
  };

  v8f acc[2][4];
  const v8f vzero = {0.f,0.f,0.f,0.f,0.f,0.f,0.f,0.f};
#pragma unroll
  for (int i = 0; i < 2; ++i)
#pragma unroll
    for (int j = 0; j < 4; ++j) acc[i][j] = vzero;

  // prologue: fill buffer 0
  stageA(0, 0);
  stageB(0, 0);
  wait_async0();
  __syncthreads();

  int cur = 0;
  for (int k0 = 0; k0 < K; k0 += BK) {
    const bool hasNext = (k0 + BK) < K;
    if (hasNext) {               // kick off DMA for next tile into the other buffer
      stageA(k0 + BK, cur ^ 1);
      stageB(k0 + BK, cur ^ 1);
    }

    // ---- fragments + 8 WMMAs per wave (32x64 wave tile, K=32 step) ----
    FragU a[2], b[4];
#pragma unroll
    for (int mt = 0; mt < 2; ++mt) {
      int r = (wm * 32 + mt * 16 + lm) * LDSROW;
      a[mt].q[0] = *(const uint4*)&As[cur][r + 8 * half];
      a[mt].q[1] = *(const uint4*)&As[cur][r + 16 + 8 * half];
    }
#pragma unroll
    for (int nt = 0; nt < 4; ++nt) {
      int r = (wn * 64 + nt * 16 + lm) * LDSROW;
      b[nt].q[0] = *(const uint4*)&Bs[cur][r + 8 * half];
      b[nt].q[1] = *(const uint4*)&Bs[cur][r + 16 + 8 * half];
    }
#pragma unroll
    for (int mt = 0; mt < 2; ++mt)
#pragma unroll
      for (int nt = 0; nt < 4; ++nt)
        acc[mt][nt] = __builtin_amdgcn_wmma_f32_16x16x32_bf16(
            false, a[mt].v, false, b[nt].v, (short)0, acc[mt][nt], false, false);

    if (hasNext) {
      wait_async0();             // own DMA into nxt has landed
      __syncthreads();           // everyone's DMA/stores into nxt visible
      cur ^= 1;
    }
  }

  // ---- epilogue: bias + scale, f32 or bf16 store ----
#pragma unroll
  for (int mt = 0; mt < 2; ++mt)
#pragma unroll
    for (int nt = 0; nt < 4; ++nt) {
      int n = n0 + wn * 64 + nt * 16 + lm;
      if (n >= N) continue;
      float bval = bias ? bias[n] : 0.0f;
#pragma unroll
      for (int i = 0; i < 8; ++i) {
        int m = m0 + wm * 32 + mt * 16 + i + 8 * half;
        float v = (acc[mt][nt][i] + bval) * scale;
        if (outF) outF[(size_t)m * ldc + n] = v;
        else      outB[(size_t)m * ldc + n] = (bf16)v;
      }
    }
}

// ---------------------------------------------------------------------------
// x[row, :] = tok_emb[idx[row], :] + pos_emb[row % T, :]  (store bf16)
// ---------------------------------------------------------------------------
__global__ __launch_bounds__(256) void embed_to_bf16(
    const int* __restrict__ idx, const float* __restrict__ tok,
    const float* __restrict__ pos, bf16* __restrict__ x, int T, int C)
{
  int row = blockIdx.x;
  int t   = row % T;
  int id  = idx[row];
  const float* tr = tok + (size_t)id * C;
  const float* pr = pos + (size_t)t  * C;
  for (int c = threadIdx.x; c < C; c += 256)
    x[(size_t)row * C + c] = (bf16)(tr[c] + pr[c]);
}

// ---------------------------------------------------------------------------
// W[K,N] f32  ->  Wt[N,K] bf16   (tile transpose through LDS)
// ---------------------------------------------------------------------------
__global__ __launch_bounds__(256) void transpose_f32_to_bf16T(
    const float* __restrict__ W, bf16* __restrict__ Wt, int K, int N)
{
  __shared__ float tile[32][33];
  int kb = blockIdx.y * 32, nb = blockIdx.x * 32;
  int tx = threadIdx.x & 31, ty = threadIdx.x >> 5;   // 32 x 8
#pragma unroll
  for (int i = 0; i < 32; i += 8) {
    int k = kb + ty + i, n = nb + tx;
    tile[ty + i][tx] = (k < K && n < N) ? W[(size_t)k * N + n] : 0.0f;
  }
  __syncthreads();
#pragma unroll
  for (int i = 0; i < 32; i += 8) {
    int n = nb + ty + i, k = kb + tx;
    if (n < N && k < K) Wt[(size_t)n * K + k] = (bf16)tile[tx][ty + i];
  }
}

// ---------------------------------------------------------------------------
// Causal softmax over row t of S [T,T] f32 -> P bf16 (zeros for s > t).
// ---------------------------------------------------------------------------
__global__ __launch_bounds__(256) void softmax_causal(
    const float* __restrict__ S, bf16* __restrict__ P, int T)
{
  int t   = blockIdx.x;
  int tid = threadIdx.x;
  const float* Sr = S + (size_t)t * T;
  __shared__ float red[256];

  float mx = -1e30f;
  for (int s = tid; s <= t; s += 256) mx = fmaxf(mx, Sr[s]);
  red[tid] = mx; __syncthreads();
  for (int o = 128; o > 0; o >>= 1) {
    if (tid < o) red[tid] = fmaxf(red[tid], red[tid + o]);
    __syncthreads();
  }
  mx = red[0]; __syncthreads();

  float sum = 0.0f;
  for (int s = tid; s <= t; s += 256) sum += __expf(Sr[s] - mx);
  red[tid] = sum; __syncthreads();
  for (int o = 128; o > 0; o >>= 1) {
    if (tid < o) red[tid] += red[tid + o];
    __syncthreads();
  }
  float inv = 1.0f / red[0];

  bf16* Pr = P + (size_t)t * T;
  for (int s = tid; s < T; s += 256)
    Pr[s] = (s <= t) ? (bf16)(__expf(Sr[s] - mx) * inv) : (bf16)0.0f;
}

// ---------------------------------------------------------------------------
// LayerNorm over last dim (C=768), eps=1e-5, store bf16.
// ---------------------------------------------------------------------------
__global__ __launch_bounds__(256) void layernorm_to_bf16(
    const float* __restrict__ y, const float* __restrict__ g,
    const float* __restrict__ b, bf16* __restrict__ out, int C)
{
  int row = blockIdx.x, tid = threadIdx.x;
  const float* yr = y + (size_t)row * C;
  float s1 = 0.f, s2 = 0.f;
  for (int c = tid; c < C; c += 256) { float v = yr[c]; s1 += v; s2 += v * v; }
  __shared__ float r1[256], r2[256];
  r1[tid] = s1; r2[tid] = s2; __syncthreads();
  for (int o = 128; o > 0; o >>= 1) {
    if (tid < o) { r1[tid] += r1[tid + o]; r2[tid] += r2[tid + o]; }
    __syncthreads();
  }
  float mu  = r1[0] / C;
  float var = r2[0] / C - mu * mu;
  float rs  = rsqrtf(var + 1e-5f);
  for (int c = tid; c < C; c += 256)
    out[(size_t)row * C + c] = (bf16)((yr[c] - mu) * rs * g[c] + b[c]);
}

// ---------------------------------------------------------------------------
extern "C" void kernel_launch(void* const* d_in, const int* in_sizes, int n_in,
                              void* d_out, int out_size, void* d_ws, size_t ws_size,
                              hipStream_t stream) {
  (void)in_sizes; (void)n_in; (void)out_size; (void)ws_size;
  const int   B = 4, T = 2048, C = 768, V = 50257;
  const int   BT = B * T;                 // 8192

  const int*   idx = (const int*)  d_in[0];
  const float* tok = (const float*)d_in[1];
  const float* pos = (const float*)d_in[2];
  const float* Wq  = (const float*)d_in[3];
  const float* bq  = (const float*)d_in[4];
  const float* Wk  = (const float*)d_in[5];
  const float* bk  = (const float*)d_in[6];
  const float* Wv  = (const float*)d_in[7];
  const float* bv  = (const float*)d_in[8];
  const float* lng = (const float*)d_in[9];
  const float* lnb = (const float*)d_in[10];
  const float* Wh  = (const float*)d_in[11];
  const float* bh  = (const float*)d_in[12];
  float* logits = (float*)d_out;

  // workspace carve-out (~194 MB total)
  char* p = (char*)d_ws;
  auto carve = [&](size_t bytes) -> char* {
    char* r = p; p += (bytes + 255) & ~(size_t)255; return r;
  };
  bf16*  xbf  = (bf16*) carve((size_t)BT * C * sizeof(bf16));
  bf16*  qbf  = (bf16*) carve((size_t)BT * C * sizeof(bf16));
  bf16*  kbf  = (bf16*) carve((size_t)BT * C * sizeof(bf16));
  bf16*  vbf  = (bf16*) carve((size_t)BT * C * sizeof(bf16));
  bf16*  WqT  = (bf16*) carve((size_t)C * C * sizeof(bf16));
  bf16*  WkT  = (bf16*) carve((size_t)C * C * sizeof(bf16));
  bf16*  WvT  = (bf16*) carve((size_t)C * C * sizeof(bf16));
  bf16*  WhT  = (bf16*) carve((size_t)V * C * sizeof(bf16));
  float* Sbuf = (float*)carve((size_t)T * T * sizeof(float));
  bf16*  Pbuf = (bf16*) carve((size_t)T * T * sizeof(bf16));
  float* ybuf = (float*)carve((size_t)BT * C * sizeof(float));
  bf16*  ynbf = (bf16*) carve((size_t)BT * C * sizeof(bf16));

  dim3 blk(256);

  // 1) weights -> bf16, transposed to [N,K]
  transpose_f32_to_bf16T<<<dim3(24, 24), blk, 0, stream>>>(Wq, WqT, C, C);
  transpose_f32_to_bf16T<<<dim3(24, 24), blk, 0, stream>>>(Wk, WkT, C, C);
  transpose_f32_to_bf16T<<<dim3(24, 24), blk, 0, stream>>>(Wv, WvT, C, C);
  transpose_f32_to_bf16T<<<dim3((V + 31) / 32, 24), blk, 0, stream>>>(Wh, WhT, C, V);

  // 2) embeddings (bf16)
  embed_to_bf16<<<BT, blk, 0, stream>>>(idx, tok, pos, xbf, T, C);

  // 3) QKV projections (q pre-scaled by 1/sqrt(C))
  const float invs = 1.0f / sqrtf((float)C);
  gemm_bf16_wmma<false><<<dim3(C / BN, BT / BM), blk, 0, stream>>>(
      xbf, WqT, bq, nullptr, qbf, BT, C, C, C, C, C, invs, 0);
  gemm_bf16_wmma<false><<<dim3(C / BN, BT / BM), blk, 0, stream>>>(
      xbf, WkT, bk, nullptr, kbf, BT, C, C, C, C, C, 1.0f, 0);
  gemm_bf16_wmma<false><<<dim3(C / BN, BT / BM), blk, 0, stream>>>(
      xbf, WvT, bv, nullptr, vbf, BT, C, C, C, C, C, 1.0f, 0);

  // 4) attention, per batch (S/P buffers reused; stream-serialized)
  for (int b = 0; b < B; ++b) {
    const bf16* qb = qbf + (size_t)b * T * C;
    const bf16* kb = kbf + (size_t)b * T * C;
    const bf16* vb = vbf + (size_t)b * T * C;
    float*      yb = ybuf + (size_t)b * T * C;
    // S = q * k^T (k already [N=T, K=C] layout), skip fully-masked blocks
    gemm_bf16_wmma<false><<<dim3(T / BN, T / BM), blk, 0, stream>>>(
        qb, kb, nullptr, Sbuf, nullptr, T, T, C, C, C, T, 1.0f, 1);
    softmax_causal<<<T, blk, 0, stream>>>(Sbuf, Pbuf, T);
    // y = P * v  (v is [K=T, N=C] row-major -> B_IS_KN path)
    gemm_bf16_wmma<true><<<dim3(C / BN, T / BM), blk, 0, stream>>>(
        Pbuf, vb, nullptr, yb, nullptr, T, C, T, T, C, C, 1.0f, 0);
  }

  // 5) LayerNorm -> bf16
  layernorm_to_bf16<<<BT, blk, 0, stream>>>(ybuf, lng, lnb, ynbf, C);

  // 6) logits = yn * Wh + bh  (f32 out, N guarded at 50257)
  gemm_bf16_wmma<false><<<dim3((V + BN - 1) / BN, BT / BM), blk, 0, stream>>>(
      ynbf, WhT, bh, logits, nullptr, BT, V, C, C, C, V, 1.0f, 0);
}
